// MoE_60911226192029
// MI455X (gfx1250) — matrix operand order repaired and checked by reading, not compiled, not run on hardware
//
#include <hip/hip_runtime.h>
#include <math.h>

#define H_DIM     1024
#define E_DIM     512
#define N_SHARED  2
#define N_ROUTED  16
#define TOP_K     4
#define T_TOK     8192
#define N_EXPERTS (N_SHARED + N_ROUTED)
#define LDA       (E_DIM + 8)   // LDS act row stride (elements); 1040B = 65*16 -> 16B aligned rows

typedef __attribute__((ext_vector_type(16))) __bf16         v16bf;
typedef __attribute__((ext_vector_type(8)))  float          v8f;
typedef __attribute__((ext_vector_type(4)))  unsigned int   v4u;
typedef __attribute__((ext_vector_type(4)))  float          v4f;
typedef __attribute__((ext_vector_type(4)))  unsigned short v4s;

union FragU { v16bf bf; v4u u[2]; };

__device__ __forceinline__ unsigned short f2bf(float f) {
  unsigned int u = __float_as_uint(f);
  unsigned int r = u + 0x7FFFu + ((u >> 16) & 1u);   // round-to-nearest-even
  return (unsigned short)(r >> 16);
}

// A-matrix fragment, 16x32 bf16 (ISA 7.12.2 16-bit A layout):
// lanes 0-15 hold rows M=0..15 with K = 0..7 (v0-3) and 16..23 (v4-7);
// lanes 16-31 hold K = 8..15 and 24..31.
__device__ __forceinline__ v16bf load_frag_a(const unsigned short* base,
                                             int row0, int k0, int ld, int lane) {
  int r = lane & 15, h = lane >> 4;
  const unsigned short* p = base + (size_t)(row0 + r) * ld + (k0 + h * 8);
  FragU f;
  f.u[0] = *(const v4u*)(p);        // K = h*8 .. h*8+7
  f.u[1] = *(const v4u*)(p + 16);   // K = h*8+16 .. h*8+23
  return f.bf;
}

// B-matrix fragment, 32x16 bf16, loaded from B^T row-major [N x K]:
// lane n (0-15) holds column n, K=0..7 and 16..23; lane n+16 holds K=8..15 and 24..31.
__device__ __forceinline__ v16bf load_frag_b(const unsigned short* __restrict__ base,
                                             int n0, int k0, int ld, int lane) {
  int r = lane & 15, h = lane >> 4;
  const unsigned short* p = base + (size_t)(n0 + r) * ld + (k0 + h * 8);
  FragU f;
  f.u[0] = *(const v4u*)(p);
  f.u[1] = *(const v4u*)(p + 16);
  return f.bf;
}

// ---------------- f32 -> bf16 conversion (vectorized x4) ----------------
__global__ void cvt_f32_bf16(const float* __restrict__ in,
                             unsigned short* __restrict__ out, int n4) {
  int i = blockIdx.x * blockDim.x + threadIdx.x;
  int stride = gridDim.x * blockDim.x;
  for (; i < n4; i += stride) {
    v4f v = *(const v4f*)(in + 4 * (size_t)i);
    v4s o;
    o.x = f2bf(v.x); o.y = f2bf(v.y); o.z = f2bf(v.z); o.w = f2bf(v.w);
    *(v4s*)(out + 4 * (size_t)i) = o;
  }
}

// ---------------- router: softmax + top-4 dense gate ----------------
__global__ __launch_bounds__(128) void router_kernel(const float* __restrict__ x,
                                                     const float* __restrict__ rw,
                                                     float* __restrict__ gate) {
  __shared__ float logits[4][N_ROUTED];
  int lane = threadIdx.x & 31;
  int wv   = threadIdx.x >> 5;
  int t    = blockIdx.x * 4 + wv;

  float xv[H_DIM / 32];
  const float* xp = x + (size_t)t * H_DIM;
#pragma unroll
  for (int i = 0; i < H_DIM / 32; ++i) xv[i] = xp[lane + 32 * i];

  for (int e = 0; e < N_ROUTED; ++e) {
    const float* wp = rw + (size_t)e * H_DIM;
    float s = 0.f;
#pragma unroll
    for (int i = 0; i < H_DIM / 32; ++i) s += xv[i] * wp[lane + 32 * i];
#pragma unroll
    for (int m = 16; m >= 1; m >>= 1) s += __shfl_xor(s, m, 32);
    if (lane == 0) logits[wv][e] = s;
  }

  if (lane == 0) {
    float mx = logits[wv][0];
    for (int e = 1; e < N_ROUTED; ++e) mx = fmaxf(mx, logits[wv][e]);
    float p[N_ROUTED];
    float sum = 0.f;
    for (int e = 0; e < N_ROUTED; ++e) { p[e] = __expf(logits[wv][e] - mx); sum += p[e]; }
    float inv = 1.f / sum;
    for (int e = 0; e < N_ROUTED; ++e) p[e] *= inv;
    unsigned sel = 0;
    for (int k = 0; k < TOP_K; ++k) {
      int bi = 0; float bv = -1.f;
      for (int e = 0; e < N_ROUTED; ++e)
        if (!((sel >> e) & 1u) && p[e] > bv) { bv = p[e]; bi = e; }
      sel |= 1u << bi;
    }
    float* g = gate + (size_t)t * N_ROUTED;
    for (int e = 0; e < N_ROUTED; ++e) g[e] = ((sel >> e) & 1u) ? p[e] : 0.f;
  }
}

// ---------------- fused MoE: up -> gelu*gate -> down, accumulated over experts ----
__global__ __launch_bounds__(256) void moe_kernel(
    const unsigned short* __restrict__ xb,
    const unsigned short* __restrict__ sup,
    const unsigned short* __restrict__ sdn,
    const unsigned short* __restrict__ rup,
    const unsigned short* __restrict__ rdn,
    const float* __restrict__ gate,
    float* __restrict__ out) {
  __shared__ unsigned short act[16 * LDA];  // 16.6 KB bf16 activations
  __shared__ float gtile[16];

  int lane = threadIdx.x & 31;
  int wv   = threadIdx.x >> 5;  // 0..7
  int t0   = blockIdx.x * 16;
  int r    = lane & 15, h = lane >> 4;

  v8f acc[8] = {};  // 16 tokens x 1024 cols, 8 16x16 tiles per wave (persistent)

  for (int e = 0; e < N_EXPERTS; ++e) {
    const unsigned short* up   = (e < N_SHARED)
        ? sup + (size_t)e * E_DIM * H_DIM
        : rup + (size_t)(e - N_SHARED) * E_DIM * H_DIM;
    const unsigned short* down = (e < N_SHARED)
        ? sdn + (size_t)e * H_DIM * E_DIM
        : rdn + (size_t)(e - N_SHARED) * H_DIM * E_DIM;

    if (threadIdx.x < 16) {
      gtile[threadIdx.x] = (e < N_SHARED)
          ? 1.0f
          : gate[(size_t)(t0 + threadIdx.x) * N_ROUTED + (e - N_SHARED)];
    }
    __syncthreads();

    // ---- phase 1: tmp[16,512] = x_tile[16,1024] @ up^T ----
    // 32 n-tiles, 4 per wave; per-tile K loop with double-buffered B so the
    // load for step k+1 is in flight while WMMA k executes.
#pragma unroll
    for (int i = 0; i < 4; ++i) {
      int n0 = (wv * 4 + i) * 16;
      v8f c = {};
      v16bf bcur = load_frag_b(up, n0, 0, H_DIM, lane);
#pragma unroll 8
      for (int kk = 0; kk < H_DIM / 32; ++kk) {
        v16bf a = load_frag_a(xb, t0, kk * 32, H_DIM, lane);
        v16bf bnext = bcur;
        if (kk + 1 < H_DIM / 32)
          bnext = load_frag_b(up, n0, (kk + 1) * 32, H_DIM, lane);
        if (kk + 2 < H_DIM / 32)
          __builtin_prefetch(up + (size_t)(n0 + r) * H_DIM + (kk + 2) * 32, 0, 0);
        c = __builtin_amdgcn_wmma_f32_16x16x32_bf16(false, a, false, bcur,
                                                    (short)0, c, false, false);
        bcur = bnext;
      }
      // exact-erf gelu * gate -> bf16 act in LDS (row-major, K contiguous)
#pragma unroll
      for (int j = 0; j < 8; ++j) {
        int m = h * 8 + j;                 // C/D layout: vgpr j, lane-half h
        float v = c[j];
        v = 0.5f * v * (1.0f + erff(v * 0.70710678118654752f));
        v *= gtile[m];
        act[m * LDA + n0 + r] = f2bf(v);
      }
    }
    __syncthreads();

    // ---- phase 2: out_tile[16,1024] += act[16,512] @ down^T ----
    // 64 n-tiles, 8 per wave; A re-read from LDS, B double-buffered.
#pragma unroll
    for (int i = 0; i < 8; ++i) {
      int n0 = (wv * 8 + i) * 16;
      v8f c = acc[i];
      v16bf bcur = load_frag_b(down, n0, 0, E_DIM, lane);
#pragma unroll
      for (int kk = 0; kk < E_DIM / 32; ++kk) {
        v16bf a = load_frag_a(act, 0, kk * 32, LDA, lane);
        v16bf bnext = bcur;
        if (kk + 1 < E_DIM / 32)
          bnext = load_frag_b(down, n0, (kk + 1) * 32, E_DIM, lane);
        if (kk + 2 < E_DIM / 32)
          __builtin_prefetch(down + (size_t)(n0 + r) * E_DIM + (kk + 2) * 32, 0, 0);
        c = __builtin_amdgcn_wmma_f32_16x16x32_bf16(false, a, false, bcur,
                                                    (short)0, c, false, false);
        bcur = bnext;
      }
      acc[i] = c;
    }
    __syncthreads();
  }

  // store f32 output tile
#pragma unroll
  for (int i = 0; i < 8; ++i) {
    int n0 = (wv * 8 + i) * 16;
#pragma unroll
    for (int j = 0; j < 8; ++j) {
      int m = h * 8 + j;
      out[(size_t)(t0 + m) * H_DIM + n0 + r] = acc[i][j];
    }
  }
}

extern "C" void kernel_launch(void* const* d_in, const int* in_sizes, int n_in,
                              void* d_out, int out_size, void* d_ws, size_t ws_size,
                              hipStream_t stream) {
  const float* x   = (const float*)d_in[0];
  const float* sup = (const float*)d_in[1];
  const float* sdn = (const float*)d_in[2];
  const float* rup = (const float*)d_in[3];
  const float* rdn = (const float*)d_in[4];
  const float* rw  = (const float*)d_in[5];
  float* out = (float*)d_out;

  const size_t nX   = (size_t)T_TOK * H_DIM;
  const size_t nSup = (size_t)N_SHARED * E_DIM * H_DIM;
  const size_t nSdn = (size_t)N_SHARED * H_DIM * E_DIM;
  const size_t nRup = (size_t)N_ROUTED * E_DIM * H_DIM;
  const size_t nRdn = (size_t)N_ROUTED * H_DIM * E_DIM;

  char* ws = (char*)d_ws;
  size_t off = 0;
  unsigned short* xb  = (unsigned short*)(ws + off); off += nX   * 2;
  unsigned short* sub = (unsigned short*)(ws + off); off += nSup * 2;
  unsigned short* sdb = (unsigned short*)(ws + off); off += nSdn * 2;
  unsigned short* rub = (unsigned short*)(ws + off); off += nRup * 2;
  unsigned short* rdb = (unsigned short*)(ws + off); off += nRdn * 2;
  float* gate = (float*)(ws + off);                  off += (size_t)T_TOK * N_ROUTED * 4;

  const int CB = 256;
  auto blocks4 = [](size_t n4) { int b = (int)((n4 + 1023) / 1024); return b > 4096 ? 4096 : b; };
  cvt_f32_bf16<<<blocks4(nX / 4),   CB, 0, stream>>>(x,   xb,  (int)(nX / 4));
  cvt_f32_bf16<<<blocks4(nSup / 4), CB, 0, stream>>>(sup, sub, (int)(nSup / 4));
  cvt_f32_bf16<<<blocks4(nSdn / 4), CB, 0, stream>>>(sdn, sdb, (int)(nSdn / 4));
  cvt_f32_bf16<<<blocks4(nRup / 4), CB, 0, stream>>>(rup, rub, (int)(nRup / 4));
  cvt_f32_bf16<<<blocks4(nRdn / 4), CB, 0, stream>>>(rdn, rdb, (int)(nRdn / 4));

  router_kernel<<<T_TOK / 4, 128, 0, stream>>>(x, rw, gate);

  moe_kernel<<<T_TOK / 16, 256, 0, stream>>>(xb, sub, sdb, rub, rdb, gate, out);
}
